// DCFormerLayer_66365834657869
// MI455X (gfx1250) — matrix operand order repaired and checked by reading, not compiled, not run on hardware
//
#include <hip/hip_runtime.h>
#include <math.h>

// ---------------------------------------------------------------------------
// Types for CDNA5 WMMA (wave32, 16x16x32 bf16 -> f32)
// ---------------------------------------------------------------------------
typedef __attribute__((ext_vector_type(16))) __bf16 v16bf;
typedef __attribute__((ext_vector_type(8)))  __bf16 v8bf;
typedef __attribute__((ext_vector_type(8)))  float  v8f;

__device__ __forceinline__ v16bf cat8(v8bf a, v8bf b) {
  return __builtin_shufflevector(a, b, 0,1,2,3,4,5,6,7,8,9,10,11,12,13,14,15);
}
__device__ __forceinline__ v8f wmma_bf16(v16bf a, v16bf b, v8f c) {
  // (neg_a, A, neg_b, B, c_mod, C, reuse_a, reuse_b)
  return __builtin_amdgcn_wmma_f32_16x16x32_bf16(false, a, false, b, (short)0, c, false, false);
}

// Problem constants
#define BB   32
#define TT   256
#define DIMD 512
#define HH   16
#define DKD  32
#define HQKV 1536
#define NROW (BB*TT)     // 8192

// ---------------------------------------------------------------------------
// Weight convert/transpose: src f32 [K][N] -> dst bf16 [N][K]
// ---------------------------------------------------------------------------
__global__ void f32_to_bf16_transpose(const float* __restrict__ src,
                                      __bf16* __restrict__ dst, int K, int N) {
  int i = blockIdx.x * blockDim.x + threadIdx.x;
  if (i >= K * N) return;
  int k = i / N, n = i % N;
  dst[(size_t)n * K + k] = (__bf16)src[i];
}

// qkw [1,4,128,4,16] f32 -> qkwT bf16 [c][i*16+m][k=128]
__global__ void qkw_convert(const float* __restrict__ src, __bf16* __restrict__ dst) {
  int i = blockIdx.x * blockDim.x + threadIdx.x;
  if (i >= 4 * 128 * 4 * 16) return;
  int m  = i & 15;
  int ii = (i >> 4) & 3;
  int kk = (i >> 6) & 127;
  int c  = (i >> 13) & 3;
  dst[((size_t)c * 64 + ii * 16 + m) * 128 + kk] = (__bf16)src[i];
}

// ---------------------------------------------------------------------------
// LayerNorm: one wave (32 lanes) per 512-wide row; output bf16
// ---------------------------------------------------------------------------
__global__ __launch_bounds__(256) void ln_kernel(const float* __restrict__ x,
                                                 const float* __restrict__ g,
                                                 const float* __restrict__ bta,
                                                 __bf16* __restrict__ xn) {
  int row  = blockIdx.x * 8 + (threadIdx.x >> 5);
  int lane = threadIdx.x & 31;
  const float* xr = x + (size_t)row * DIMD;
  float v[16], s = 0.f, s2 = 0.f;
#pragma unroll
  for (int i = 0; i < 16; ++i) { v[i] = xr[lane + i * 32]; s += v[i]; s2 += v[i] * v[i]; }
#pragma unroll
  for (int m = 1; m < 32; m <<= 1) { s += __shfl_xor(s, m, 32); s2 += __shfl_xor(s2, m, 32); }
  float mu  = s * (1.f / DIMD);
  float var = s2 * (1.f / DIMD) - mu * mu;
  float rs  = rsqrtf(var + 1e-5f);
#pragma unroll
  for (int i = 0; i < 16; ++i) {
    int c = lane + i * 32;
    xn[(size_t)row * DIMD + c] = (__bf16)((v[i] - mu) * rs * g[c] + bta[c]);
  }
}

// ---------------------------------------------------------------------------
// Register-tiled bf16 WMMA GEMM: C[M,N] = act(A[M,K] @ Bt[N,K]^T + bias)
// Each wave computes a (MT*16) x (NT*16) tile: MT A-frags x NT B-frags per
// 32-wide k-step combined in MT*NT WMMAs (register-level A/B reuse).
// 4 waves per block tile the N direction. act: 0 none, 1 gelu, 2 tanh.
// ---------------------------------------------------------------------------
template <int MT, int NT>
__global__ __launch_bounds__(128) void gemm_bf16_wmma(
    const __bf16* __restrict__ A, int lda,
    const __bf16* __restrict__ Bt, int ldb,
    void* __restrict__ C, int ldc, int c_is_f32,
    int M, int N, int K,
    const float* __restrict__ bias, int act) {
  const int lane = threadIdx.x & 31;
  const int wv   = threadIdx.x >> 5;
  const int nt0  = (blockIdx.x * 4 + wv) * NT;   // in 16-col units
  const int mt0  = blockIdx.y * MT;              // in 16-row units
  if (nt0 * 16 >= N || mt0 * 16 >= M) return;
  const int ln = lane & 15;
  const int hf = (lane >> 4) & 1;

  const __bf16* ap[MT];
#pragma unroll
  for (int mi = 0; mi < MT; ++mi)
    ap[mi] = A + (size_t)((mt0 + mi) * 16 + ln) * lda + hf * 8;
  const __bf16* bp[NT];
#pragma unroll
  for (int ni = 0; ni < NT; ++ni)
    bp[ni] = Bt + (size_t)((nt0 + ni) * 16 + ln) * ldb + hf * 8;

  v8f acc[MT][NT];
#pragma unroll
  for (int mi = 0; mi < MT; ++mi)
#pragma unroll
    for (int ni = 0; ni < NT; ++ni)
      acc[mi][ni] = (v8f){0.f,0.f,0.f,0.f,0.f,0.f,0.f,0.f};

  for (int kk = 0; kk < K; kk += 32) {
    v16bf af[MT];
#pragma unroll
    for (int mi = 0; mi < MT; ++mi)
      af[mi] = cat8(*(const v8bf*)(ap[mi] + kk), *(const v8bf*)(ap[mi] + kk + 16));
    v16bf bfv[NT];
#pragma unroll
    for (int ni = 0; ni < NT; ++ni)
      bfv[ni] = cat8(*(const v8bf*)(bp[ni] + kk), *(const v8bf*)(bp[ni] + kk + 16));
#pragma unroll
    for (int mi = 0; mi < MT; ++mi)
#pragma unroll
      for (int ni = 0; ni < NT; ++ni)
        acc[mi][ni] = wmma_bf16(af[mi], bfv[ni], acc[mi][ni]);
  }

#pragma unroll
  for (int mi = 0; mi < MT; ++mi) {
#pragma unroll
    for (int ni = 0; ni < NT; ++ni) {
      const int col = (nt0 + ni) * 16 + ln;
      const float bv = bias ? bias[col] : 0.f;
#pragma unroll
      for (int r = 0; r < 8; ++r) {
        int row = (mt0 + mi) * 16 + r + hf * 8;
        float v = acc[mi][ni][r] + bv;
        if (act == 1)      v = 0.5f * v * (1.f + erff(v * 0.70710678118654752f));
        else if (act == 2) v = tanhf(v);
        if (c_is_f32) ((float*)C)[(size_t)row * ldc + col] = v;
        else          ((__bf16*)C)[(size_t)row * ldc + col] = (__bf16)v;
      }
    }
  }
}

// ---------------------------------------------------------------------------
// q/k RMS-norm + scale, v transpose; one thread per (row, head)
//   q_bf[row][h*32+d]   = rms(q)*q_scale           (also the qv feature vector)
//   k_bf[row][h*32+d]   = rms(k)*k_scale*DK^-0.5   (attn scale folded in)
//   vT [b][h][d][t]     = v                        ([B,H,DK,T] bf16)
// ---------------------------------------------------------------------------
__global__ void qkvnorm_kernel(const float* __restrict__ qkv,
                               const float* __restrict__ qs,
                               const float* __restrict__ ks,
                               __bf16* __restrict__ qb,
                               __bf16* __restrict__ kb,
                               __bf16* __restrict__ vT) {
  int i = blockIdx.x * blockDim.x + threadIdx.x;
  if (i >= NROW * HH) return;
  int h = i & 15, row = i >> 4;
  int b = row >> 8, t = row & 255;
  const float* base = qkv + (size_t)row * HQKV + h * DKD;
  float qv[DKD], kv[DKD], sq = 0.f, sk = 0.f;
#pragma unroll
  for (int d = 0; d < DKD; ++d) {
    qv[d] = base[d]; kv[d] = base[512 + d];
    sq += qv[d] * qv[d]; sk += kv[d] * kv[d];
  }
  float rq = rsqrtf(sq * (1.f / DKD) + 1e-6f);
  float rk = rsqrtf(sk * (1.f / DKD) + 1e-6f) * 0.17677669529663687f; // *DK^-0.5
#pragma unroll
  for (int d = 0; d < DKD; ++d) {
    qb[(size_t)row * 512 + h * 32 + d] = (__bf16)(qv[d] * rq * qs[d]);
    kb[(size_t)row * 512 + h * 32 + d] = (__bf16)(kv[d] * rk * ks[d]);
    vT[(((size_t)b * HH + h) * DKD + d) * TT + t] = (__bf16)base[1024 + d];
  }
}

// RMS-norm w1 (i=0,1 of wmix[row][c][i][16]) over the 16 heads, in place
__global__ void w1rms_kernel(float* __restrict__ wmix) {
  int i = blockIdx.x * blockDim.x + threadIdx.x;
  if (i >= NROW * 4 * 2) return;
  int ii = i & 1, c = (i >> 1) & 3, row = i >> 3;
  float* p = wmix + (size_t)row * 256 + c * 64 + ii * 16;
  float v[16], s = 0.f;
#pragma unroll
  for (int j = 0; j < 16; ++j) { v[j] = p[j]; s += v[j] * v[j]; }
  float rs = rsqrtf(s * (1.f / 16.f) + 1e-6f);
#pragma unroll
  for (int j = 0; j < 16; ++j) p[j] = v[j] * rs;
}

// ---------------------------------------------------------------------------
// Fused attention: grid=(T/16, B), block=512 (16 waves, one head per wave).
// Logits tile [16t x 256s] lives in 16 v8f WMMA accumulators per wave.
// All strided gathers use loop-invariant base pointers + immediate offsets
// (per-s-tile strides: k 16KB, wmix 16KB, dd 4KB, idxs 64B) so the unrolled
// loops lower to clauses of offset loads with no per-iteration address math.
// LDS: HQ[2][16][256]f32 + HK[2][16][256]f32 (64KB, cross-head reductions)
//      PR[16][16][256]bf16 (128KB, probs staged for the P@V WMMA)
// ---------------------------------------------------------------------------
__global__ __launch_bounds__(512, 1) void attn_kernel(
    const __bf16* __restrict__ q, const __bf16* __restrict__ k,
    const __bf16* __restrict__ vT,
    const float* __restrict__ wmix, const float* __restrict__ dd,
    const float* __restrict__ biases, const int* __restrict__ idxs, int n_off,
    __bf16* __restrict__ o) {
  extern __shared__ char smem[];
  float* HQ = (float*)smem;                 // 2*16*256 floats
  float* HK = HQ + 2 * 16 * 256;            // 2*16*256 floats
  __bf16* PR = (__bf16*)(HK + 2 * 16 * 256);// 16*16*256 bf16

  const int b  = blockIdx.y;
  const int tt = blockIdx.x;
  const int h    = threadIdx.x >> 5;
  const int lane = threadIdx.x & 31;
  const int ln = lane & 15;
  const int hf = (lane >> 4) & 1;
  const int t0 = tt * 16;

  // ---- logits = q @ k^T (scale folded into k), 16 s-tiles ----
  v8f acc[16];
  {
    const __bf16* qp = q + ((size_t)(b * TT + t0 + ln) * HH + h) * DKD;
    v16bf afrag = cat8(*(const v8bf*)(qp + hf * 8), *(const v8bf*)(qp + 16 + hf * 8));
    // per-s-tile stride: 16 rows * 512 bf16 = 8192 elements (16KB immediate)
    const __bf16* kbase = k + ((size_t)(b * TT + ln) * HH + h) * DKD + hf * 8;
#pragma unroll
    for (int st = 0; st < 16; ++st) {
      const __bf16* kp = kbase + st * (16 * HH * DKD);
      v8f z = {0.f,0.f,0.f,0.f,0.f,0.f,0.f,0.f};
      acc[st] = wmma_bf16(afrag, cat8(*(const v8bf*)kp, *(const v8bf*)(kp + 16)), z);
    }
  }

  // ---- relative-position bias (idxs row base per r, +64B per s-tile) ----
  {
    const float* bb = biases + (size_t)h * n_off;
#pragma unroll
    for (int r = 0; r < 8; ++r) {
      const int* ib = idxs + (t0 + r + hf * 8) * TT + ln;
#pragma unroll
      for (int st = 0; st < 16; ++st)
        acc[st][r] += bb[ib[st * 16]];
    }
  }

  // ---- cross-head mixing (rank-2 over heads + diagonal), used twice ----
  auto mix = [&](int cq, int ck, int ddq_off, int ddk_off) {
    for (int i = threadIdx.x; i < 4 * 16 * 256; i += 512) HQ[i] = 0.f; // HQ+HK
    __syncthreads();
    // per-row (t) scalars: base + tl*256 floats
    float qw1v[8][2], qw2v[8][2], qddv[8];
    {
      const float* wq = wmix + (size_t)(b * TT + t0 + hf * 8) * 256 + cq * 64 + h;
      const float* dq = dd + (size_t)(b * TT + t0 + hf * 8) * 64 + ddq_off + h;
#pragma unroll
      for (int r = 0; r < 8; ++r) {
        qw1v[r][0] = wq[r * 256 + 0];
        qw1v[r][1] = wq[r * 256 + 16];
        qw2v[r][0] = wq[r * 256 + 32];
        qw2v[r][1] = wq[r * 256 + 48];
        qddv[r]    = dq[r * 64];
      }
    }
    // per-s scalars: base + st*16*256 floats (16KB immediate stride)
    const float* wk = wmix + (size_t)(b * TT + ln) * 256 + ck * 64 + h;
    const float* dk = dd + (size_t)(b * TT + ln) * 64 + ddk_off + h;
    const int ls = ln;                 // LDS column for this lane
    const int tlb = hf * 8;            // row-half base
#pragma unroll
    for (int st = 0; st < 16; ++st) {
      float kw1a = wk[st * 4096 + 0];
      float kw1b = wk[st * 4096 + 16];
#pragma unroll
      for (int r = 0; r < 8; ++r) {
        int tl = r + tlb;
        float x = acc[st][r];
        atomicAdd(&HQ[tl * 256 + ls + st * 16], x * qw1v[r][0]);
        atomicAdd(&HQ[(16 + tl) * 256 + ls + st * 16], x * qw1v[r][1]);
        atomicAdd(&HK[tl * 256 + ls + st * 16], x * kw1a);
        atomicAdd(&HK[(16 + tl) * 256 + ls + st * 16], x * kw1b);
      }
    }
    __syncthreads();
#pragma unroll
    for (int st = 0; st < 16; ++st) {
      float kw2a = wk[st * 4096 + 32];
      float kw2b = wk[st * 4096 + 48];
      float kddv = dk[st * 1024];
#pragma unroll
      for (int r = 0; r < 8; ++r) {
        int tl = r + tlb;
        float x = acc[st][r];
        acc[st][r] = x
          + qw2v[r][0] * HQ[tl * 256 + ls + st * 16]
          + qw2v[r][1] * HQ[(16 + tl) * 256 + ls + st * 16]
          + kw2a * HK[tl * 256 + ls + st * 16]
          + kw2b * HK[(16 + tl) * 256 + ls + st * 16]
          + x * (qddv[r] + kddv);
      }
    }
    __syncthreads(); // planes reused by next mix
  };

  mix(0, 1, 0, 16);   // pre: qw c=0, kw c=1, pre_qdd dd[0:16], pre_kdd dd[16:32]

  // ---- softmax over s (rows t=r and t=r+8 live in separate lane halves) ----
#pragma unroll
  for (int r = 0; r < 8; ++r) {
    float mx = -3.4e38f;
#pragma unroll
    for (int st = 0; st < 16; ++st) mx = fmaxf(mx, acc[st][r]);
#pragma unroll
    for (int m = 1; m < 16; m <<= 1) mx = fmaxf(mx, __shfl_xor(mx, m, 32));
    float sum = 0.f;
#pragma unroll
    for (int st = 0; st < 16; ++st) { float e = __expf(acc[st][r] - mx); acc[st][r] = e; sum += e; }
#pragma unroll
    for (int m = 1; m < 16; m <<= 1) sum += __shfl_xor(sum, m, 32);
    float inv = 1.f / sum;
#pragma unroll
    for (int st = 0; st < 16; ++st) acc[st][r] *= inv;
  }

  mix(2, 3, 32, 48);  // post: qw c=2, kw c=3, post_qdd dd[32:48], post_kdd dd[48:64]

  // ---- stage probs as bf16 rows [16t][256s] for A-fragment loads ----
  __bf16* myP = PR + (size_t)h * 16 * 256;
  {
    __bf16* pw = myP + hf * 8 * 256 + ln;
#pragma unroll
    for (int st = 0; st < 16; ++st)
#pragma unroll
      for (int r = 0; r < 8; ++r)
        pw[r * 256 + st * 16] = (__bf16)acc[st][r];
  }
  __syncthreads();

  // ---- O = P @ V : K=256 (8 WMMA steps), N=32 (2 d-tiles) ----
  const __bf16* apb = myP + ln * 256 + hf * 8;
#pragma unroll
  for (int dt = 0; dt < 2; ++dt) {
    v8f oa = {0.f,0.f,0.f,0.f,0.f,0.f,0.f,0.f};
    const __bf16* bpb = vT + (((size_t)(b * HH + h)) * DKD + dt * 16 + ln) * TT + hf * 8;
#pragma unroll
    for (int kkk = 0; kkk < 8; ++kkk) {
      v16bf af = cat8(*(const v8bf*)(apb + kkk * 32), *(const v8bf*)(apb + kkk * 32 + 16));
      v16bf bf = cat8(*(const v8bf*)(bpb + kkk * 32), *(const v8bf*)(bpb + kkk * 32 + 16));
      oa = wmma_bf16(af, bf, oa);
    }
    __bf16* ob = o + (size_t)(b * TT + t0 + hf * 8) * 512 + h * 32 + dt * 16 + ln;
#pragma unroll
    for (int r = 0; r < 8; ++r)
      ob[r * 512] = (__bf16)oa[r];
  }
}

// ---------------------------------------------------------------------------
// Workspace layout (bytes)
// ---------------------------------------------------------------------------
static const size_t OFF_XN     = 0;                        // 8192*512*2
static const size_t OFF_WQKVT  = OFF_XN     + 8388608;     // 1536*512*2
static const size_t OFF_QBF    = OFF_WQKVT  + 1572864;     // 8192*512*2
static const size_t OFF_KBF    = OFF_QBF    + 8388608;
static const size_t OFF_VT     = OFF_KBF    + 8388608;
static const size_t OFF_DW1T   = OFF_VT     + 8388608;     // 512*512*2
static const size_t OFF_QKWT   = OFF_DW1T   + 524288;      // 4*64*128*2
static const size_t OFF_DDWT   = OFF_QKWT   + 65536;       // 64*512*2
static const size_t OFF_WPROJT = OFF_DDWT   + 65536;       // 512*512*2
static const size_t OFF_BIG    = OFF_WPROJT + 524288;      // 8192*1536*4 (qkv f32)
// overlaid on BIG once qkv f32 is consumed:
static const size_t OFF_DWH    = OFF_BIG;                  // 8192*512*2 bf16
static const size_t OFF_WMIX   = OFF_BIG    + 8388608;     // 8192*256*4 f32
static const size_t OFF_DD     = OFF_WMIX   + 8388608;     // 8192*64*4  f32
static const size_t OFF_OBF    = OFF_DD     + 2097152;     // 8192*512*2 bf16

extern "C" void kernel_launch(void* const* d_in, const int* in_sizes, int n_in,
                              void* d_out, int out_size, void* d_ws, size_t ws_size,
                              hipStream_t stream) {
  const float* x       = (const float*)d_in[0];
  const float* ln_g    = (const float*)d_in[1];
  const float* ln_b    = (const float*)d_in[2];
  const float* w_qkv   = (const float*)d_in[3];
  const float* b_qkv   = (const float*)d_in[4];
  const float* q_scale = (const float*)d_in[5];
  const float* k_scale = (const float*)d_in[6];
  const float* dw1     = (const float*)d_in[7];
  const float* qkw     = (const float*)d_in[8];
  const float* dd_w    = (const float*)d_in[9];
  const float* biases  = (const float*)d_in[10];
  const float* w_proj  = (const float*)d_in[11];
  const float* b_proj  = (const float*)d_in[12];
  const int*   idxs    = (const int*)d_in[13];
  const int n_off = in_sizes[10] / HH;

  char* ws = (char*)d_ws;
  __bf16* xn     = (__bf16*)(ws + OFF_XN);
  __bf16* wqkvT  = (__bf16*)(ws + OFF_WQKVT);
  __bf16* q_bf   = (__bf16*)(ws + OFF_QBF);
  __bf16* k_bf   = (__bf16*)(ws + OFF_KBF);
  __bf16* vT     = (__bf16*)(ws + OFF_VT);
  __bf16* dw1T   = (__bf16*)(ws + OFF_DW1T);
  __bf16* qkwT   = (__bf16*)(ws + OFF_QKWT);
  __bf16* ddwT   = (__bf16*)(ws + OFF_DDWT);
  __bf16* wprojT = (__bf16*)(ws + OFF_WPROJT);
  float*  qkvf   = (float*)(ws + OFF_BIG);
  __bf16* dwh    = (__bf16*)(ws + OFF_DWH);
  float*  wmix   = (float*)(ws + OFF_WMIX);
  float*  ddv    = (float*)(ws + OFF_DD);
  __bf16* o_bf   = (__bf16*)(ws + OFF_OBF);

  // allow 192KB dynamic LDS for the fused attention kernel
  hipFuncSetAttribute((const void*)attn_kernel,
                      hipFuncAttributeMaxDynamicSharedMemorySize, 196608);

  // weight converts (transpose to [N][K] bf16 for contiguous B-fragment loads)
  f32_to_bf16_transpose<<<(512 * 1536 + 255) / 256, 256, 0, stream>>>(w_qkv, wqkvT, 512, 1536);
  f32_to_bf16_transpose<<<(512 * 512 + 255) / 256, 256, 0, stream>>>(dw1, dw1T, 512, 512);
  f32_to_bf16_transpose<<<(512 * 64 + 255) / 256, 256, 0, stream>>>(dd_w, ddwT, 512, 64);
  f32_to_bf16_transpose<<<(512 * 512 + 255) / 256, 256, 0, stream>>>(w_proj, wprojT, 512, 512);
  qkw_convert<<<(32768 + 255) / 256, 256, 0, stream>>>(qkw, qkwT);

  // LayerNorm
  ln_kernel<<<NROW / 8, 256, 0, stream>>>(x, ln_g, ln_b, xn);

  // QKV GEMM: [8192,512] @ [512,1536] -> f32  (32x64 tile per wave)
  gemm_bf16_wmma<2, 4><<<dim3(HQKV / 256, NROW / 32), 128, 0, stream>>>(
      xn, 512, wqkvT, 512, qkvf, HQKV, 1, NROW, HQKV, 512, b_qkv, 0);

  // q/k RMS norm + scales, v transpose
  qkvnorm_kernel<<<(NROW * HH) / 256, 256, 0, stream>>>(qkvf, q_scale, k_scale,
                                                        q_bf, k_bf, vT);

  // dw_hidden = gelu(qv @ dw1) : [8192,512]@[512,512] -> bf16
  gemm_bf16_wmma<2, 4><<<dim3(512 / 256, NROW / 32), 128, 0, stream>>>(
      q_bf, 512, dw1T, 512, dwh, 512, 0, NROW, 512, 512, nullptr, 1);

  // grouped w = dw_hidden @ qkw : 4x ([8192,128]@[128,64]) -> wmix f32
  for (int c = 0; c < 4; ++c) {
    gemm_bf16_wmma<2, 1><<<dim3(1, NROW / 32), 128, 0, stream>>>(
        dwh + c * 128, 512, qkwT + (size_t)c * 64 * 128, 128,
        wmix + c * 64, 256, 1, NROW, 64, 128, nullptr, 0);
  }
  // RMS-norm w1 over heads
  w1rms_kernel<<<(NROW * 8) / 256, 256, 0, stream>>>(wmix);

  // dd = tanh(qv @ dd_w) : [8192,512]@[512,64] -> f32
  gemm_bf16_wmma<2, 1><<<dim3(1, NROW / 32), 128, 0, stream>>>(
      q_bf, 512, ddwT, 512, ddv, 64, 1, NROW, 64, 512, nullptr, 2);

  // fused attention (logits+bias+mix+softmax+mix+P@V)
  attn_kernel<<<dim3(TT / 16, BB), 512, 196608, stream>>>(
      q_bf, k_bf, vT, wmix, ddv, biases, idxs, n_off, o_bf);

  // output projection: [8192,512]@[512,512] + b_proj -> d_out f32
  gemm_bf16_wmma<2, 4><<<dim3(512 / 256, NROW / 32), 128, 0, stream>>>(
      o_bf, 512, wprojT, 512, d_out, 512, 1, NROW, 512, 512, b_proj, 0);
}
